// FasterRCNN_73650099192093
// MI455X (gfx1250) — compile-verified, matrix operand order checked by reading, not run on hardware
//
#include <hip/hip_runtime.h>
#include <hip/hip_bf16.h>

// ---------------------------------------------------------------------------
// Types
// ---------------------------------------------------------------------------
typedef __attribute__((ext_vector_type(16))) _Float16 v16h;
typedef __attribute__((ext_vector_type(8)))  _Float16 v8h;
typedef __attribute__((ext_vector_type(8)))  float    v8f;
typedef __attribute__((ext_vector_type(4)))  unsigned int u32x4;
typedef __attribute__((ext_vector_type(8)))  unsigned int u32x8;

#define BM 128
#define BN 64
#define BK 32
#define LDP 40    // padded LDS row stride (halfs) for the A tile
#define LDPB 72   // TDM-padded LDS row stride (halfs) for the B tile: 64 + 8

// ---------------------------------------------------------------------------
// TDM: DMA one 32(K) x 64(N) f16 weight tile into LDS with padded rows.
// D# per CDNA5 ISA.  All inputs are wave-uniform.
//   group0: count=1 | lds_addr | global_addr[56:0] | type=2
//   group1: data_size=2B, pad_enable, pad_interval=32 DWORDs (code 4),
//           pad_amount=4 DWORDs (code 3) -> 144B LDS row stride,
//           tensor_dim0=remN, tensor_dim1=remK (OOB reads return zero),
//           tile_dim0=64, tile_dim1=32, tensor_dim0_stride=Cout.
// ---------------------------------------------------------------------------
__device__ __forceinline__ void tdm_load_b_tile(const _Float16* gsrc,
                                                unsigned lds_addr,
                                                int remN, int remK, int Cout)
{
    const unsigned long long ga = (unsigned long long)gsrc;
    u32x4 g0;
    g0[0] = 1u;                                                // count=1 (valid)
    g0[1] = lds_addr;                                          // LDS byte address
    g0[2] = (unsigned)(ga & 0xFFFFFFFFu);                      // global_addr lo
    g0[3] = (unsigned)((ga >> 32) & 0x01FFFFFFu) | (2u << 30); // hi + type=2
    u32x8 g1;
    g1[0] = (1u << 16) | (1u << 20) | (4u << 22) | (3u << 25); // 2B, pad 32dw+4dw
    g1[1] = ((unsigned)remN & 0xFFFFu) << 16;                  // tensor_dim0 lo16
    g1[2] = (((unsigned)remN >> 16) & 0xFFFFu) |               // tensor_dim0 hi16
            (((unsigned)remK & 0xFFFFu) << 16);                // tensor_dim1 lo16
    g1[3] = (((unsigned)remK >> 16) & 0xFFFFu) | (64u << 16);  // tile_dim0=64
    g1[4] = 32u;                                               // tile_dim1=32
    g1[5] = (unsigned)Cout;                                    // dim0 stride lo32
    g1[6] = 0u;                                                // stride hi / dim1 stride
    g1[7] = 0u;
    asm volatile("tensor_load_to_lds %0, %1" :: "s"(g0), "s"(g1) : "memory");
}

// ---------------------------------------------------------------------------
// Implicit-GEMM conv, NHWC activations, f16 WMMA, double-buffered LDS.
// X: [H][W][Cin] f16; Wm: [K][Cout] f16 with k = tap*Cin + ci; Y: [H][W][Cout].
// 256 threads = 8 wave32; each wave computes a 2x2 grid of 16x16 subtiles
// (4 v_wmma per K-chunk).  A tile: register-staged aligned b128 global loads
// (Cin%32==0 => a 32-aligned K-chunk never straddles a 3x3 tap).  B tile:
// Tensor Data Mover DMA straight into padded LDS, completion via TENSORcnt.
// ---------------------------------------------------------------------------
__global__ __launch_bounds__(256) void conv_wmma_kernel(
    const _Float16* __restrict__ X,
    const _Float16* __restrict__ Wm,
    const float*    __restrict__ bias,
    _Float16*       __restrict__ Y,
    int Cin, int Cout, int H, int W, int ks, int relu)
{
    const int K   = Cin * ks * ks;
    const int M   = H * W;
    const int m0  = blockIdx.x * BM;
    const int n0  = blockIdx.y * BN;
    const int pad = ks >> 1;

    __shared__ __align__(16) _Float16 As[2][BM][LDP];    // 2 x 10 KB
    __shared__ __align__(16) _Float16 Bs[2][BK][LDPB];   // 2 x 4.5 KB (TDM dest)

    const int t    = threadIdx.x;
    const int wave = t >> 5;
    const int lane = t & 31;
    const int g    = lane >> 4;        // half-wave group
    const int l    = lane & 15;

    const int wm = wave & 3;           // M-subtile pair: rows wm*32 .. wm*32+31
    const int wn = (wave >> 2) * 2;    // N-subtiles wn, wn+1

    v8f acc[2][2] = { { v8f{}, v8f{} }, { v8f{}, v8f{} } };

    // A staging: 256 thr -> 128 rows x 2 halves of the 32-wide K-chunk
    const int arow = t & 127;
    const int aq   = (t >> 7) * 16;            // k_local base: 0 or 16
    const int am   = m0 + arow;
    const int aoy  = am / W;
    const int aox  = am - aoy * W;
    const bool fastC = ((Cin & 31) == 0);

    auto loadA = [&](int k0, v8h& r0, v8h& r1) {
        r0 = v8h{}; r1 = v8h{};
        const int kb = k0 + aq;
        if (fastC) {
            // whole 16-half run shares one tap; contiguous & 16B aligned in NHWC
            const int tap = kb / Cin;
            const int cib = kb - tap * Cin;
            const int dy  = tap / ks - pad;
            const int dx  = tap - (tap / ks) * ks - pad;
            const int iy  = aoy + dy, ix = aox + dx;
            if (am < M && iy >= 0 && iy < H && ix >= 0 && ix < W) {
                const _Float16* src = X + ((size_t)iy * W + ix) * Cin + cib;
                r0 = *(const v8h*)src;
                r1 = *(const v8h*)(src + 8);
            }
        } else {
            #pragma unroll
            for (int i = 0; i < 16; ++i) {
                const int k = kb + i;
                if (k < K && am < M) {
                    const int tap = k / Cin;
                    const int ci  = k - tap * Cin;
                    const int dy  = tap / ks - pad;
                    const int dx  = tap % ks - pad;
                    const int iy  = aoy + dy, ix = aox + dx;
                    if (iy >= 0 && iy < H && ix >= 0 && ix < W) {
                        const _Float16 v = X[((size_t)iy * W + ix) * Cin + ci];
                        if (i < 8) r0[i] = v; else r1[i - 8] = v;
                    }
                }
            }
        }
    };
    auto storeA = [&](int p, const v8h& a0, const v8h& a1) {
        *(v8h*)&As[p][arow][aq]     = a0;
        *(v8h*)&As[p][arow][aq + 8] = a1;
    };
    auto issueB = [&](int k0, int buf) {   // wave-uniform TDM issue
        tdm_load_b_tile(Wm + (size_t)k0 * Cout + n0,
                        (unsigned)(unsigned long long)&Bs[buf][0][0],
                        Cout - n0, K - k0, Cout);
    };

    // ---- prologue: stage chunk 0 into buffer 0 -------------------------------
    if (wave == 0) issueB(0, 0);
    {
        v8h a0, a1;
        loadA(0, a0, a1);
        storeA(0, a0, a1);
    }
    if (wave == 0) __builtin_amdgcn_s_wait_tensorcnt(0);
    __syncthreads();

    int p = 0;
    for (int k0 = 0; k0 < K; k0 += BK) {
        const bool hasNext = (k0 + BK) < K;
        v8h na0, na1;
        if (hasNext) {                       // overlap next-chunk staging w/ WMMA
            if (wave == 0) issueB(k0 + BK, p ^ 1);
            loadA(k0 + BK, na0, na1);
        }
        if (k0 + 2 * BK < K)                 // gfx1250 global_prefetch_b8
            __builtin_prefetch(&Wm[(size_t)(k0 + 2 * BK) * Cout + n0 + (t & 63)], 0, 1);

        // ---- A fragments: lane l = row, group g picks K halves
        //      {g*8..+7, 16+g*8..+7} (two ds_load_b128 each) --------------------
        v16h afr[2], bfr[2];
        #pragma unroll
        for (int im = 0; im < 2; ++im) {
            const int r = (wm * 2 + im) * 16 + l;
            const v8h lo = *(const v8h*)&As[p][r][g * 8];
            const v8h hi = *(const v8h*)&As[p][r][16 + g * 8];
            #pragma unroll
            for (int i = 0; i < 8; ++i) { afr[im][i] = lo[i]; afr[im][i + 8] = hi[i]; }
        }
        // ---- B fragments from TDM's [k][n] layout: lane l = column,
        //      conflict-free (lanes hit consecutive banks per k) ----------------
        #pragma unroll
        for (int jn = 0; jn < 2; ++jn) {
            const int n = (wn + jn) * 16 + l;
            #pragma unroll
            for (int i = 0; i < 8; ++i) {
                bfr[jn][i]     = Bs[p][g * 8 + i][n];
                bfr[jn][i + 8] = Bs[p][16 + g * 8 + i][n];
            }
        }
        #pragma unroll
        for (int im = 0; im < 2; ++im)
            #pragma unroll
            for (int jn = 0; jn < 2; ++jn)
                acc[im][jn] = __builtin_amdgcn_wmma_f32_16x16x32_f16(
                    false, afr[im], false, bfr[jn], (short)0, acc[im][jn], false, false);

        if (hasNext) {
            storeA(p ^ 1, na0, na1);
            if (wave == 0) __builtin_amdgcn_s_wait_tensorcnt(0);
        }
        __syncthreads();
        p ^= 1;
    }

    // ---- epilogue: C/D layout VGPR v -> M = v + 8*g, N = l ------------------
    #pragma unroll
    for (int im = 0; im < 2; ++im) {
        #pragma unroll
        for (int jn = 0; jn < 2; ++jn) {
            const int n = n0 + (wn + jn) * 16 + l;
            if (n >= Cout) continue;
            const float bz = bias ? bias[n] : 0.0f;
            #pragma unroll
            for (int v = 0; v < 8; ++v) {
                const int mm = m0 + (wm * 2 + im) * 16 + v + g * 8;
                if (mm >= M) continue;
                float o = acc[im][jn][v] + bz;
                if (relu) o = fmaxf(o, 0.0f);
                Y[(size_t)mm * Cout + n] = (_Float16)o;   // NHWC store
            }
        }
    }
}

// ---------------------------------------------------------------------------
// Helpers: NCHW f32 -> NHWC f16, weight transpose, NHWC 2x2 maxpool
// ---------------------------------------------------------------------------
__global__ void img_to_nhwc_kernel(const float* __restrict__ x,   // [3][HW]
                                   _Float16* __restrict__ y,      // [HW][3]
                                   int HW)
{
    int i = blockIdx.x * 256 + threadIdx.x;
    if (i < HW * 3) {
        const int sp = i / 3, c = i - sp * 3;
        y[i] = (_Float16)x[(size_t)c * HW + sp];
    }
}

// Wsrc: [Cout][Cin][ks2] f32 (OIHW), Wm: [tap*Cin+ci][Cout] f16
__global__ void wtranspose_kernel(const float* __restrict__ Wsrc,
                                  _Float16* __restrict__ Wm,
                                  int Cout, int Cin, int ks2)
{
    const int K = Cin * ks2;
    int i = blockIdx.x * 256 + threadIdx.x;
    if (i < Cout * K) {
        const int k   = i / Cout;
        const int n   = i - k * Cout;
        const int tap = k / Cin;
        const int ci  = k - tap * Cin;
        Wm[i] = (_Float16)Wsrc[(size_t)n * K + (size_t)ci * ks2 + tap];
    }
}

__global__ void maxpool2_kernel(const _Float16* __restrict__ x,  // [H][W][C]
                                _Float16* __restrict__ y,        // [H/2][W/2][C]
                                int C, int H, int W)
{
    const int Ho = H >> 1, Wo = W >> 1;
    int i = blockIdx.x * 256 + threadIdx.x;
    if (i >= Ho * Wo * C) return;
    const int c  = i % C;
    const int sp = i / C;
    const int yo = sp / Wo, xo = sp - yo * Wo;
    const _Float16* p = x + (((size_t)(yo * 2) * W) + xo * 2) * C + c;
    const size_t rs = (size_t)W * C;
    const float a = (float)p[0],  b = (float)p[C];
    const float d = (float)p[rs], e = (float)p[rs + C];
    y[i] = (_Float16)fmaxf(fmaxf(a, b), fmaxf(d, e));
}

// ---------------------------------------------------------------------------
// RPN post-processing (cls/reg are NHWC == reference transpose(0,2,3,1))
// ---------------------------------------------------------------------------
#define FEAT 32
#define NA   9
#define NANC (FEAT * FEAT * NA)   // 9216
#define PRE_NMS 2000

__global__ void rpn_decode_kernel(const _Float16* __restrict__ cls,  // [1024][18]
                                  const _Float16* __restrict__ reg,  // [1024][36]
                                  float* __restrict__ scores,
                                  float* __restrict__ boxes)
{
    int n = blockIdx.x * 256 + threadIdx.x;
    if (n >= NANC) return;
    const int a  = n % NA;
    const int sp = n / NA;
    const int yy = sp / FEAT, xx = sp - yy * FEAT;

    const float c0 = (float)cls[(size_t)sp * 18 + a * 2 + 0];
    const float c1 = (float)cls[(size_t)sp * 18 + a * 2 + 1];
    const float mx = fmaxf(c0, c1);
    const float e0 = __expf(c0 - mx), e1 = __expf(c1 - mx);
    const float fg = e1 / (e0 + e1);

    const float dx = (float)reg[(size_t)sp * 36 + a * 4 + 0];
    const float dy = (float)reg[(size_t)sp * 36 + a * 4 + 1];
    float dw = (float)reg[(size_t)sp * 36 + a * 4 + 2];
    float dh = (float)reg[(size_t)sp * 36 + a * 4 + 3];
    dw = fminf(fmaxf(dw, -4.0f), 4.0f);
    dh = fminf(fmaxf(dh, -4.0f), 4.0f);

    const float ratios[3] = {0.5f, 1.0f, 2.0f};
    const float scales[3] = {8.0f, 16.0f, 32.0f};
    const float sz = scales[a % 3] * 16.0f;
    const float rt = ratios[a / 3];
    const float aw = sz * sqrtf(1.0f / rt);
    const float ah = sz * sqrtf(rt);
    const float cxa = ((float)xx + 0.5f) * 16.0f;
    const float cya = ((float)yy + 0.5f) * 16.0f;

    const float cx = cxa + dx * aw;
    const float cy = cya + dy * ah;
    const float w  = aw * __expf(dw);
    const float h  = ah * __expf(dh);

    scores[n] = fg;
    boxes[4 * n + 0] = fminf(fmaxf(cx - 0.5f * w, 0.0f), 512.0f);
    boxes[4 * n + 1] = fminf(fmaxf(cy - 0.5f * h, 0.0f), 512.0f);
    boxes[4 * n + 2] = fminf(fmaxf(cx + 0.5f * w, 0.0f), 512.0f);
    boxes[4 * n + 3] = fminf(fmaxf(cy + 0.5f * h, 0.0f), 512.0f);
}

// Exact top-k via rank counting (ties broken by lower index, matching top_k).
__global__ __launch_bounds__(256) void topk_rank_kernel(
    const float* __restrict__ scores, const float* __restrict__ boxes,
    float* __restrict__ ss, float* __restrict__ bs)
{
    __shared__ float sc[NANC];   // 36 KB of the 320 KB WGP LDS
    for (int i = threadIdx.x; i < NANC; i += 256) sc[i] = scores[i];
    __syncthreads();

    const int i = blockIdx.x * 256 + threadIdx.x;
    if (i >= NANC) return;
    const float si = sc[i];
    int rank = 0;
    for (int j = 0; j < NANC; ++j) {
        const float sj = sc[j];
        rank += (sj > si) || (sj == si && j < i);
    }
    if (rank < PRE_NMS) {
        ss[rank] = si;
        bs[4 * rank + 0] = boxes[4 * i + 0];
        bs[4 * rank + 1] = boxes[4 * i + 1];
        bs[4 * rank + 2] = boxes[4 * i + 2];
        bs[4 * rank + 3] = boxes[4 * i + 3];
    }
}

// Single-block greedy NMS over the 2000 sorted candidates + masked output.
__global__ __launch_bounds__(256) void nms_write_kernel(
    const float* __restrict__ bs, const float* __restrict__ ss,
    float* __restrict__ out)
{
    __shared__ float bx[PRE_NMS][4];          // 32 KB
    __shared__ float area[PRE_NMS];           //  8 KB
    __shared__ unsigned char keep[PRE_NMS];   //  2 KB

    for (int i = threadIdx.x; i < PRE_NMS; i += 256) {
        const float x0 = bs[4 * i + 0], y0 = bs[4 * i + 1];
        const float x1 = bs[4 * i + 2], y1 = bs[4 * i + 3];
        bx[i][0] = x0; bx[i][1] = y0; bx[i][2] = x1; bx[i][3] = y1;
        area[i] = (x1 - x0) * (y1 - y0);
        keep[i] = 1;
    }
    __syncthreads();

    for (int i = 0; i < PRE_NMS; ++i) {
        if (keep[i]) {   // uniform branch: keep[] is shared
            const float xi0 = bx[i][0], yi0 = bx[i][1];
            const float xi1 = bx[i][2], yi1 = bx[i][3];
            const float ai  = area[i];
            for (int j = i + 1 + (int)threadIdx.x; j < PRE_NMS; j += 256) {
                if (!keep[j]) continue;
                const float lx = fmaxf(xi0, bx[j][0]);
                const float ly = fmaxf(yi0, bx[j][1]);
                const float rx = fminf(xi1, bx[j][2]);
                const float ry = fminf(yi1, bx[j][3]);
                const float iw = fmaxf(rx - lx, 0.0f);
                const float ih = fmaxf(ry - ly, 0.0f);
                const float inter = iw * ih;
                const float iou = inter / (ai + area[j] - inter + 1e-6f);
                if (iou > 0.7f) keep[j] = 0;
            }
        }
        __syncthreads();
    }

    for (int i = threadIdx.x; i < PRE_NMS; i += 256) {
        const float k = keep[i] ? 1.0f : 0.0f;
        out[4 * i + 0] = bx[i][0] * k;
        out[4 * i + 1] = bx[i][1] * k;
        out[4 * i + 2] = bx[i][2] * k;
        out[4 * i + 3] = bx[i][3] * k;
        out[4 * PRE_NMS + i] = ss[i] * k;
    }
}

// ---------------------------------------------------------------------------
// Host orchestration (graph-capture safe: only stream launches, ws scratch)
// ---------------------------------------------------------------------------
extern "C" void kernel_launch(void* const* d_in, const int* in_sizes, int n_in,
                              void* d_out, int out_size, void* d_ws, size_t ws_size,
                              hipStream_t stream)
{
    (void)in_sizes; (void)n_in; (void)out_size; (void)ws_size;

    const float* img = (const float*)d_in[0];
    const float* Wb[13]; const float* bb[13];
    for (int i = 0; i < 13; ++i) {
        Wb[i] = (const float*)d_in[1 + 2 * i];
        bb[i] = (const float*)d_in[2 + 2 * i];
    }
    const float* W3 = (const float*)d_in[27];
    const float* b3 = (const float*)d_in[28];
    const float* Wc = (const float*)d_in[29];
    const float* bc = (const float*)d_in[30];
    const float* Wr = (const float*)d_in[31];
    const float* br = (const float*)d_in[32];

    char* ws = (char*)d_ws;
    size_t off = 0;
    auto take = [&](size_t bytes) -> char* {
        char* p = ws + off;
        off += (bytes + 255) & ~(size_t)255;
        return p;
    };
    _Float16* actA = (_Float16*)take(64ull * 512 * 512 * 2);   // 32 MB ping
    _Float16* actB = (_Float16*)take(64ull * 512 * 512 * 2);   // 32 MB pong
    _Float16* wmat = (_Float16*)take(4608ull * 512 * 2);       // max [K][Cout] f16
    _Float16* imgh = (_Float16*)take(3ull * 512 * 512 * 2);
    _Float16* clsb = (_Float16*)take(18ull * 1024 * 2);
    _Float16* regb = (_Float16*)take(36ull * 1024 * 2);
    float* scores  = (float*)take((size_t)NANC * 4);
    float* boxes   = (float*)take((size_t)NANC * 16);
    float* ssort   = (float*)take((size_t)PRE_NMS * 4);
    float* bsort   = (float*)take((size_t)PRE_NMS * 16);

    auto run_conv = [&](const _Float16* src, _Float16* dst,
                        const float* Wf, const float* bf,
                        int cin, int cout, int H, int W, int ks, int relu) {
        const int K  = cin * ks * ks;
        const int tw = K * cout;
        wtranspose_kernel<<<(tw + 255) / 256, 256, 0, stream>>>(Wf, wmat, cout, cin, ks * ks);
        dim3 grid((H * W + BM - 1) / BM, (cout + BN - 1) / BN);
        conv_wmma_kernel<<<grid, 256, 0, stream>>>(src, wmat, bf, dst,
                                                   cin, cout, H, W, ks, relu);
    };
    auto run_pool = [&](const _Float16* src, _Float16* dst, int C, int H, int W) {
        const int n = C * (H / 2) * (W / 2);
        maxpool2_kernel<<<(n + 255) / 256, 256, 0, stream>>>(src, dst, C, H, W);
    };

    // image NCHW f32 -> NHWC f16
    img_to_nhwc_kernel<<<(3 * 512 * 512 + 255) / 256, 256, 0, stream>>>(img, imgh, 512 * 512);

    // VGG-16 backbone (CFG = 64,64,M,128,128,M,256,256,256,M,512,512,512,M,512,512,512)
    run_conv(imgh, actA, Wb[0],  bb[0],   3,  64, 512, 512, 3, 1);
    run_conv(actA, actB, Wb[1],  bb[1],  64,  64, 512, 512, 3, 1);
    run_pool(actB, actA, 64, 512, 512);
    run_conv(actA, actB, Wb[2],  bb[2],  64, 128, 256, 256, 3, 1);
    run_conv(actB, actA, Wb[3],  bb[3], 128, 128, 256, 256, 3, 1);
    run_pool(actA, actB, 128, 256, 256);
    run_conv(actB, actA, Wb[4],  bb[4], 128, 256, 128, 128, 3, 1);
    run_conv(actA, actB, Wb[5],  bb[5], 256, 256, 128, 128, 3, 1);
    run_conv(actB, actA, Wb[6],  bb[6], 256, 256, 128, 128, 3, 1);
    run_pool(actA, actB, 256, 128, 128);
    run_conv(actB, actA, Wb[7],  bb[7], 256, 512,  64,  64, 3, 1);
    run_conv(actA, actB, Wb[8],  bb[8], 512, 512,  64,  64, 3, 1);
    run_conv(actB, actA, Wb[9],  bb[9], 512, 512,  64,  64, 3, 1);
    run_pool(actA, actB, 512, 64, 64);
    run_conv(actB, actA, Wb[10], bb[10], 512, 512, 32, 32, 3, 1);
    run_conv(actA, actB, Wb[11], bb[11], 512, 512, 32, 32, 3, 1);
    run_conv(actB, actA, Wb[12], bb[12], 512, 512, 32, 32, 3, 1);
    // feat in actA

    // RPN head
    run_conv(actA, actB, W3, b3, 512, 512, 32, 32, 3, 1);   // h in actB
    run_conv(actB, clsb, Wc, bc, 512,  18, 32, 32, 1, 0);
    run_conv(actB, regb, Wr, br, 512,  36, 32, 32, 1, 0);

    // softmax + anchor decode -> exact top-2000 by rank -> greedy NMS -> out
    rpn_decode_kernel<<<(NANC + 255) / 256, 256, 0, stream>>>(clsb, regb, scores, boxes);
    topk_rank_kernel<<<(NANC + 255) / 256, 256, 0, stream>>>(scores, boxes, ssort, bsort);
    nms_write_kernel<<<1, 256, 0, stream>>>(bsort, ssort, (float*)d_out);
}
